// GCN_2_4097398800599
// MI455X (gfx1250) — matrix operand order repaired and checked by reading, not compiled, not run on hardware
//
#include <hip/hip_runtime.h>
#include <hip/hip_bf16.h>
#include <stdint.h>

#define D 128
#define NLAYERS 5

typedef __attribute__((ext_vector_type(16))) __bf16 v16bf;
typedef __attribute__((ext_vector_type(8)))  float  v8f;

union FragU { uint4 q[2]; v16bf v; };

// round-to-nearest-even f32 -> bf16 (raw bits)
__device__ __forceinline__ unsigned short f2bf(float f) {
  unsigned u = __float_as_uint(f);
  u += 0x7FFFu + ((u >> 16) & 1u);
  return (unsigned short)(u >> 16);
}

// monotone IEEE-f32 -> u32 encoding: enc(a) < enc(b)  <=>  a < b
// enc of any finite float is >= 0x00800000, so 0 is a safe "-inf/no edge" sentinel.
__device__ __forceinline__ unsigned enc_f32(float f) {
  unsigned u = __float_as_uint(f);
  return (u & 0x80000000u) ? ~u : (u | 0x80000000u);
}
__device__ __forceinline__ float dec_f32(unsigned u) {
  unsigned b = (u & 0x80000000u) ? (u & 0x7FFFFFFFu) : ~u;
  return __uint_as_float(b);
}

__global__ void cvt_bf16_kernel(const float* __restrict__ in,
                                unsigned short* __restrict__ out, int n) {
  int i = blockIdx.x * blockDim.x + threadIdx.x;
  if (i < n) out[i] = f2bf(in[i]);
}

// h[m,n] = sum_k xb[m,k] * wb[n,k] + bias[n]
// one wave -> one 16x16 tile; block = 256 threads = 8 waves = 8 n-tiles of one m-tile
__global__ void gemm_wmma_kernel(const unsigned short* __restrict__ xb,
                                 const unsigned short* __restrict__ wb,
                                 const float* __restrict__ bias,
                                 float* __restrict__ h, int M) {
  const int lane = threadIdx.x & 31;
  const int wave = threadIdx.x >> 5;
  const int half = lane >> 4;
  const int l16  = lane & 15;
  const int m0 = blockIdx.x * 16;
  const int n0 = wave * 16;
  if (m0 >= M) return;

  const unsigned short* arow = xb + (size_t)(m0 + l16) * D;  // A row m=l16 (both lane halves)
  const unsigned short* brow = wb + (size_t)(n0 + l16) * D;  // B col n=l16 = W row

  v8f c = {};
#pragma unroll
  for (int k0 = 0; k0 < D; k0 += 32) {
    FragU a, b;
    // A frag: lane holds K = k0 + half*8 + [0..7]  and  k0 + 16 + half*8 + [0..7]
    a.q[0] = *(const uint4*)(arow + k0 + half * 8);
    a.q[1] = *(const uint4*)(arow + k0 + 16 + half * 8);
    // B frag: lane holds K = k0 + half*8 + [0..7]  and  k0 + 16 + half*8 + [0..7]
    b.q[0] = *(const uint4*)(brow + k0 + half * 8);
    b.q[1] = *(const uint4*)(brow + k0 + 16 + half * 8);
    c = __builtin_amdgcn_wmma_f32_16x16x32_bf16(false, a.v, false, b.v,
                                                (short)0, c, false, false);
  }

  const float bv = bias[n0 + l16];
#pragma unroll
  for (int r = 0; r < 8; ++r) {
    // C/D layout: VGPR r -> row r + 8*half, col l16
    h[(size_t)(m0 + 8 * half + r) * D + n0 + l16] = c[r] + bv;
  }
}

__global__ void init_m_kernel(unsigned* __restrict__ m, int n) {
  int i = blockIdx.x * blockDim.x + threadIdx.x;
  if (i < n) m[i] = 0u;  // sentinel: below enc of any finite float
}

// one wave per edge: lane handles 4 features (b128 gather + 4 u32 atomic max)
__global__ void scatter_max_kernel(const float* __restrict__ h,
                                   const int* __restrict__ src,
                                   const int* __restrict__ dst,
                                   unsigned* __restrict__ m, int E) {
  int gid = blockIdx.x * blockDim.x + threadIdx.x;
  int e = gid >> 5;
  int lane = gid & 31;
  if (e >= E) return;
  int s = src[e], d = dst[e];
  const float4 v = *(const float4*)(h + (size_t)s * D + lane * 4);
  unsigned* mp = m + (size_t)d * D + lane * 4;
  atomicMax(mp + 0, enc_f32(v.x));
  atomicMax(mp + 1, enc_f32(v.y));
  atomicMax(mp + 2, enc_f32(v.z));
  atomicMax(mp + 3, enc_f32(v.w));
}

// hidden layers: decode, no-edge -> 0, SiLU, re-quantize to bf16 for next GEMM
__global__ void act_silu_bf16_kernel(const unsigned* __restrict__ m,
                                     unsigned short* __restrict__ xb, int n) {
  int i = blockIdx.x * blockDim.x + threadIdx.x;
  if (i < n) {
    unsigned u = m[i];
    float f = (u == 0u) ? 0.0f : dec_f32(u);
    float s = f / (1.0f + __expf(-f));
    xb[i] = f2bf(s);
  }
}

// last layer: decode straight to f32 output
__global__ void final_out_kernel(const unsigned* __restrict__ m,
                                 float* __restrict__ out, int n) {
  int i = blockIdx.x * blockDim.x + threadIdx.x;
  if (i < n) {
    unsigned u = m[i];
    out[i] = (u == 0u) ? 0.0f : dec_f32(u);
  }
}

extern "C" void kernel_launch(void* const* d_in, const int* in_sizes, int n_in,
                              void* d_out, int out_size, void* d_ws, size_t ws_size,
                              hipStream_t stream) {
  const float* x  = (const float*)d_in[0];
  const int*   ei = (const int*)d_in[1];
  const float* Ws = (const float*)d_in[2];
  const float* bs = (const float*)d_in[3];

  const int ND = in_sizes[0];      // N * D
  const int N  = ND / D;
  const int E  = in_sizes[1] / 2;
  const int* src = ei;
  const int* dst = ei + E;

  // workspace layout (all offsets 256B aligned; total ~64.2 MB)
  char* ws = (char*)d_ws;
  size_t offWb = 0;
  size_t szWb  = (size_t)NLAYERS * D * D * sizeof(unsigned short);
  size_t offXb = (offWb + szWb + 255) & ~(size_t)255;
  size_t szXb  = (size_t)ND * sizeof(unsigned short);
  size_t offH  = (offXb + szXb + 255) & ~(size_t)255;
  size_t szH   = (size_t)ND * sizeof(float);
  size_t offM  = (offH + szH + 255) & ~(size_t)255;

  unsigned short* Wb   = (unsigned short*)(ws + offWb);
  unsigned short* Xb   = (unsigned short*)(ws + offXb);
  float*          H    = (float*)(ws + offH);
  unsigned*       Menc = (unsigned*)(ws + offM);

  const int T = 256;
  const int nw = NLAYERS * D * D;
  cvt_bf16_kernel<<<(nw + T - 1) / T, T, 0, stream>>>(Ws, Wb, nw);
  cvt_bf16_kernel<<<(ND + T - 1) / T, T, 0, stream>>>(x, Xb, ND);

  const int mtiles = (N + 15) / 16;                 // 3125
  const int scatter_blocks = (int)(((size_t)E * 32 + T - 1) / T);  // 100000

  for (int layer = 0; layer < NLAYERS; ++layer) {
    gemm_wmma_kernel<<<mtiles, 256, 0, stream>>>(
        Xb, Wb + (size_t)layer * D * D, bs + (size_t)layer * D, H, N);
    init_m_kernel<<<(ND + T - 1) / T, T, 0, stream>>>(Menc, ND);
    scatter_max_kernel<<<scatter_blocks, T, 0, stream>>>(H, src, dst, Menc, E);
    if (layer < NLAYERS - 1)
      act_silu_bf16_kernel<<<(ND + T - 1) / T, T, 0, stream>>>(Menc, Xb, ND);
    else
      final_out_kernel<<<(ND + T - 1) / T, T, 0, stream>>>(Menc, (float*)d_out, ND);
  }
}
